// DeepseekMLA_55662776156808
// MI455X (gfx1250) — matrix operand order repaired and checked by reading, not compiled, not run on hardware
//
#include <hip/hip_runtime.h>
#include <hip/hip_bf16.h>
#include <math.h>
#include <stdint.h>

// ---------------------------------------------------------------------------
// Types
// ---------------------------------------------------------------------------
typedef __bf16 bf16_t;
typedef __attribute__((ext_vector_type(16))) __bf16 v16bf;
typedef __attribute__((ext_vector_type(8)))  float  v8f;
typedef unsigned int v4u __attribute__((ext_vector_type(4)));
typedef int          v4i __attribute__((ext_vector_type(4)));
typedef int          v8i __attribute__((ext_vector_type(8)));

#define T_SEQ  2048
#define HID    5120
#define NH     16
#define QL     1536
#define KL     512
#define NOPE   128
#define ROPE_D 64
#define VD     128
#define QK     192   // NOPE + ROPE

union AFragU { v16bf v; float4 f4[2]; };
union BCvt4 { double d;  bf16_t b[4]; };

#ifndef __has_builtin
#define __has_builtin(x) 0
#endif
#if __has_builtin(__builtin_amdgcn_tensor_load_to_lds) && \
    __has_builtin(__builtin_amdgcn_s_wait_tensorcnt)
#define HAVE_TDM 1
#else
#define HAVE_TDM 0
#endif

// ---------------------------------------------------------------------------
// CDNA5 async copy helpers (ASYNCcnt path, ISA 08_async_tensor §4)
// ---------------------------------------------------------------------------
__device__ __forceinline__ void async_load_b128(unsigned lds_addr,
                                                const void* gaddr) {
  // LDS[lds_addr .. +15] = MEM[gaddr .. +15], tracked with ASYNCcnt
  asm volatile("global_load_async_to_lds_b128 %0, %1, off"
               :: "v"(lds_addr), "v"((unsigned long long)gaddr)
               : "memory");
}
__device__ __forceinline__ void wait_async0() {
  asm volatile("s_wait_asynccnt 0x0" ::: "memory");
}
// Generic shared pointer -> wave-relative LDS byte address (aperture rule:
// the low 32 bits of the generic address).
__device__ __forceinline__ unsigned lds_off(const void* p) {
  return (unsigned)(uintptr_t)p;
}

// ---------------------------------------------------------------------------
// Tensor Data Mover: 2D tile load Global -> LDS (TENSORcnt path, ISA 08 §7-8)
// D# group0: count=1 | lds_addr | global_addr[56:0] | type=2
// D# group1: data_size=2B | tensor_dim0/1 | tile_dim0/1 | tensor_dim0_stride
// Groups 2/3 zero (2D tensor). Issued once per wave; EXEC ignored.
// ---------------------------------------------------------------------------
#if HAVE_TDM
__device__ __forceinline__ void tdm_load_2d(unsigned lds_addr,
                                            const void* gaddr,
                                            unsigned tile_k, unsigned tile_rows,
                                            unsigned tensor_k,
                                            unsigned tensor_rows,
                                            unsigned row_stride_elems) {
  unsigned long long ga = (unsigned long long)gaddr;
  v4u g0;
  g0[0] = 1u;                                          // count=1 (valid), user
  g0[1] = lds_addr;                                    // lds_addr[31:0]
  g0[2] = (unsigned)(ga & 0xffffffffu);                // global_addr[31:0]
  g0[3] = (unsigned)((ga >> 32) & 0x01ffffffu)         // global_addr[56:32]
          | (2u << 30);                                // type = 2 ("image")
  v8i g1;
  g1[0] = (int)(1u << 16);                             // data_size=1 -> 2 bytes
  g1[1] = (int)((tensor_k & 0xffffu) << 16);           // tensor_dim0[15:0]
  g1[2] = (int)(((tensor_k >> 16) & 0xffffu) |
                ((tensor_rows & 0xffffu) << 16));      // dim0[31:16]|dim1[15:0]
  g1[3] = (int)(((tensor_rows >> 16) & 0xffffu) |
                ((tile_k & 0xffffu) << 16));           // dim1[31:16]|tile_dim0
  g1[4] = (int)(tile_rows & 0xffffu);                  // tile_dim1 (tile_dim2=0)
  g1[5] = (int)row_stride_elems;                       // tensor_dim0_stride[31:0]
  g1[6] = 0;                                           // stride[47:32], dim1_stride lo
  g1[7] = 0;
  v4i z4 = {0, 0, 0, 0};
#if defined(__clang_major__) && (__clang_major__ >= 23)
  v8i z8 = {0, 0, 0, 0, 0, 0, 0, 0};
  __builtin_amdgcn_tensor_load_to_lds(g0, g1, z4, z4, z8, 0);
#else
  __builtin_amdgcn_tensor_load_to_lds(g0, g1, z4, z4, 0);
#endif
}
__device__ __forceinline__ void wait_tensor0() {
  __builtin_amdgcn_s_wait_tensorcnt(0);
}
#endif

// ---------------------------------------------------------------------------
// YaRN inverse frequency (matches reference _yarn_inv_freq)
// ---------------------------------------------------------------------------
static __device__ __forceinline__ float yarn_inv_freq(int j) {
  float pf    = __powf(10000.0f, (2.0f * (float)j) / 64.0f);
  float extra = 1.0f / pf;
  float inter = 1.0f / (40.0f * pf);
  const float two_pi = 6.283185307179586f;
  float cd_fast = 64.0f * __logf(4096.0f / (32.0f * two_pi)) / (2.0f * __logf(10000.0f));
  float cd_slow = 64.0f * __logf(4096.0f / ( 1.0f * two_pi)) / (2.0f * __logf(10000.0f));
  float lo = fmaxf(floorf(cd_fast), 0.0f);
  float hi = fminf(ceilf(cd_slow), 63.0f);
  float ramp = ((float)j - lo) / fmaxf(hi - lo, 0.001f);
  ramp = fminf(fmaxf(ramp, 0.0f), 1.0f);
  return inter * ramp + extra * (1.0f - ramp);
}

// ---------------------------------------------------------------------------
// f32 -> bf16 conversion (4 elements / thread)
// ---------------------------------------------------------------------------
__global__ __launch_bounds__(256) void cvt_bf16_kernel(
    const float* __restrict__ x, bf16_t* __restrict__ y, long n4) {
  long i = (long)blockIdx.x * 256 + threadIdx.x;
  if (i >= n4) return;
  float4 f = ((const float4*)x)[i];
  BCvt4 u;
  u.b[0] = (bf16_t)f.x; u.b[1] = (bf16_t)f.y;
  u.b[2] = (bf16_t)f.z; u.b[3] = (bf16_t)f.w;
  ((double*)y)[i] = u.d;
}

// ---------------------------------------------------------------------------
// Transposing f32 -> bf16 weight conversion: y(N x K) = x(K x N)^T
// 32x32 tiles through LDS, coalesced on both sides. K,N multiples of 32.
// ---------------------------------------------------------------------------
__global__ __launch_bounds__(256) void cvt_t_bf16_kernel(
    const float* __restrict__ x, bf16_t* __restrict__ y, int Kdim, int Ndim) {
  __shared__ float tile[32][33];
  const int kb = blockIdx.y * 32, nb = blockIdx.x * 32;
  const int tx = threadIdx.x & 31, ty = threadIdx.x >> 5;  // 8 rows / pass
#pragma unroll
  for (int p = 0; p < 4; ++p) {
    int k = kb + ty + p * 8;
    tile[ty + p * 8][tx] = x[(size_t)k * Ndim + nb + tx];
  }
  __syncthreads();
#pragma unroll
  for (int p = 0; p < 4; ++p) {
    int n = nb + ty + p * 8;
    y[(size_t)n * Kdim + kb + tx] = (bf16_t)tile[tx][ty + p * 8];
  }
}

// ---------------------------------------------------------------------------
// bf16 GEMM with transposed B:
//   C(MxN f32) = A(MxK bf16 rowmajor) * Bt(NxK bf16 rowmajor)^T
// Block = 128 threads (4 waves), 64x64 tile, K-step 32, double-buffered LDS.
// A tile staged by the Tensor Data Mover (wave 0, TENSORcnt); B tile staged by
// per-thread global_load_async_to_lds_b128 (ASYNCcnt).
// ---------------------------------------------------------------------------
__global__ __launch_bounds__(128) void gemm_bf16_kernel(
    const bf16_t* __restrict__ A, const bf16_t* __restrict__ Bt,
    float* __restrict__ C, int M, int N, int K) {
  __shared__ bf16_t As[2][64 * 32];   // row-major [m][k]
  __shared__ bf16_t Bs[2][64 * 32];   // row-major [n][k]
  const int tid  = threadIdx.x;
  const int wave = tid >> 5;
  const int lane = tid & 31;
  const int hi   = lane >> 4;
  const int l16  = lane & 15;
  const int m0   = blockIdx.y * 64;
  const int n0   = blockIdx.x * 64;
  // async staging coords: thread covers rows r0 and r0+32, one 8-elem chunk
  const int r0 = tid >> 2;
  const int c0 = (tid & 3) << 3;

  auto stage = [&](int buf, int k0) {
#if HAVE_TDM
    if (wave == 0) {
      tdm_load_2d(lds_off(&As[buf][0]), A + (size_t)m0 * K + k0,
                  /*tile_k=*/32, /*tile_rows=*/64,
                  /*tensor_k=*/(unsigned)K, /*tensor_rows=*/64,
                  /*row_stride=*/(unsigned)K);
    }
#else
    async_load_b128(lds_off(&As[buf][r0 * 32 + c0]),
                    A + (size_t)(m0 + r0) * K + k0 + c0);
    async_load_b128(lds_off(&As[buf][(r0 + 32) * 32 + c0]),
                    A + (size_t)(m0 + r0 + 32) * K + k0 + c0);
#endif
    async_load_b128(lds_off(&Bs[buf][r0 * 32 + c0]),
                    Bt + (size_t)(n0 + r0) * K + k0 + c0);
    async_load_b128(lds_off(&Bs[buf][(r0 + 32) * 32 + c0]),
                    Bt + (size_t)(n0 + r0 + 32) * K + k0 + c0);
  };

  v8f acc[4] = {};
  stage(0, 0);
  int buf = 0;
  for (int k0 = 0; k0 < K; k0 += 32) {
    wait_async0();        // my async B-tile stores have landed
#if HAVE_TDM
    wait_tensor0();       // wave 0's TDM A-tile has landed (others: TENSORcnt==0)
#endif
    __syncthreads();      // everyone's have landed; prev buffer fully consumed
    if (k0 + 32 < K) stage(buf ^ 1, k0 + 32);

    // A fragment (16x32), ISA 16-bit A striping
    AFragU a;
    a.f4[0] = *(const float4*)&As[buf][(wave * 16 + l16) * 32 + hi * 8];
    a.f4[1] = *(const float4*)&As[buf][(wave * 16 + l16) * 32 + 16 + hi * 8];
#pragma unroll
    for (int j = 0; j < 4; ++j) {
      // B fragment (32x16): lane = column n, K chunks hi*8 and 16+hi*8 from Bs
      AFragU bu;
      bu.f4[0] = *(const float4*)&Bs[buf][(j * 16 + l16) * 32 + hi * 8];
      bu.f4[1] = *(const float4*)&Bs[buf][(j * 16 + l16) * 32 + 16 + hi * 8];
      v16bf b = bu.v;
      acc[j] = __builtin_amdgcn_wmma_f32_16x16x32_bf16(
          false, a.v, false, b, (short)0, acc[j], false, false);
    }
    buf ^= 1;
  }

  // epilogue: C layout VGPR r -> row r (+8 for upper lanes)
#pragma unroll
  for (int j = 0; j < 4; ++j) {
    int n = n0 + j * 16 + l16;
#pragma unroll
    for (int r = 0; r < 8; ++r) {
      int m = m0 + wave * 16 + r + hi * 8;
      C[(size_t)m * N + n] = acc[j][r];
    }
  }
}

// ---------------------------------------------------------------------------
// RMS norm: Y(bf16) = X * rsqrt(mean(X^2)+eps) * W  (one row per block)
// ---------------------------------------------------------------------------
__global__ __launch_bounds__(256) void rmsnorm_bf16_kernel(
    const float* __restrict__ X, const float* __restrict__ W,
    bf16_t* __restrict__ Y, int C, int ldx) {
  __shared__ float red[8];
  const int row = blockIdx.x;
  const int tid = threadIdx.x;
  const float* x = X + (size_t)row * ldx;
  float ss = 0.f;
  for (int c = tid; c < C; c += 256) { float v = x[c]; ss += v * v; }
#pragma unroll
  for (int off = 16; off >= 1; off >>= 1) ss += __shfl_xor(ss, off, 32);
  if ((tid & 31) == 0) red[tid >> 5] = ss;
  __syncthreads();
  if (tid == 0) {
    float s = 0.f;
#pragma unroll
    for (int w = 0; w < 8; ++w) s += red[w];
    red[0] = rsqrtf(s / (float)C + 1e-6f);
  }
  __syncthreads();
  float r = red[0];
  for (int c = tid; c < C; c += 256)
    Y[(size_t)row * C + c] = (bf16_t)(x[c] * r * W[c]);
}

// ---------------------------------------------------------------------------
// Build Q: apply RoPE to q_pe, fold attn scale, emit bf16 (T,H,192)
// ---------------------------------------------------------------------------
__global__ __launch_bounds__(256) void build_q_kernel(
    const int* __restrict__ pos, const float* __restrict__ Qf,
    bf16_t* __restrict__ Qout, float attn_scale, float csm) {
  __shared__ float cs[32], sn[32];
  const int t = blockIdx.x, tid = threadIdx.x;
  if (tid < 32) {
    float f = (float)pos[t] * yarn_inv_freq(tid);
    cs[tid] = __cosf(f) * csm;
    sn[tid] = __sinf(f) * csm;
  }
  __syncthreads();
  const float* q = Qf + (size_t)t * (NH * QK);
  bf16_t* o = Qout + (size_t)t * (NH * QK);
  for (int idx = tid; idx < NH * QK; idx += 256) {
    int h = idx / QK, d = idx - h * QK;
    float val;
    if (d < NOPE) {
      val = q[idx];
    } else {
      int j = (d - NOPE) >> 1;
      float x1 = q[h * QK + NOPE + 2 * j];
      float x2 = q[h * QK + NOPE + 2 * j + 1];
      val = ((d & 1) == 0) ? (x1 * cs[j] - x2 * sn[j])
                           : (x2 * cs[j] + x1 * sn[j]);
    }
    o[idx] = (bf16_t)(val * attn_scale);
  }
}

// ---------------------------------------------------------------------------
// Build K (T,H,192 bf16: k_nope | roped k_pe broadcast) and V transposed
// (H,VD,VLD bf16) so P*V B-fragments are contiguous along the K(t) dim.
// ---------------------------------------------------------------------------
__global__ __launch_bounds__(256) void build_k_kernel(
    const int* __restrict__ pos, const float* __restrict__ lat,
    const float* __restrict__ kvf, bf16_t* __restrict__ Kout,
    bf16_t* __restrict__ Vt, float csm, int vld) {
  __shared__ float cs[32], sn[32], pe[64];
  const int t = blockIdx.x, tid = threadIdx.x;
  if (tid < 32) {
    float f = (float)pos[t] * yarn_inv_freq(tid);
    cs[tid] = __cosf(f) * csm;
    sn[tid] = __sinf(f) * csm;
  }
  __syncthreads();
  const float* lrow = lat + (size_t)t * (KL + ROPE_D);
  if (tid < 64) {
    int j = tid >> 1;
    float x1 = lrow[KL + 2 * j];
    float x2 = lrow[KL + 2 * j + 1];
    pe[tid] = ((tid & 1) == 0) ? (x1 * cs[j] - x2 * sn[j])
                               : (x2 * cs[j] + x1 * sn[j]);
  }
  __syncthreads();
  const float* kvrow = kvf + (size_t)t * (NH * (NOPE + VD));
  bf16_t* ko = Kout + (size_t)t * (NH * QK);
  for (int idx = tid; idx < NH * QK; idx += 256) {
    int h = idx / QK, d = idx - h * QK;
    float val = (d < NOPE) ? kvrow[h * (NOPE + VD) + d] : pe[d - NOPE];
    ko[idx] = (bf16_t)val;
  }
  for (int idx = tid; idx < NH * VD; idx += 256) {
    int h = idx / VD, d = idx - h * VD;
    Vt[((size_t)h * VD + d) * vld + t] =
        (bf16_t)kvrow[h * (NOPE + VD) + NOPE + d];
  }
}

// ---------------------------------------------------------------------------
// Flash attention: one wave per (16-row q tile, head). Streams K in chunks of
// 32 with online softmax. Scores: 12x wmma (K=192). P*V: 8x wmma after a
// C-layout -> A-layout re-stripe through LDS.
// ---------------------------------------------------------------------------
__global__ __launch_bounds__(32) void mla_attn_kernel(
    const bf16_t* __restrict__ Q, const bf16_t* __restrict__ Kf,
    const bf16_t* __restrict__ Vt, bf16_t* __restrict__ O, int vld) {
  __shared__ bf16_t Ps[16 * 32];
  const int lane = threadIdx.x;
  const int hi   = lane >> 4;
  const int l16  = lane & 15;
  const int t0   = blockIdx.x * 16;
  const int h    = blockIdx.y;

  // Q fragments for the whole 16x192 row block (stay resident).
  AFragU qa[6];
  {
    const bf16_t* qrow = Q + ((size_t)(t0 + l16) * NH + h) * QK;
#pragma unroll
    for (int f = 0; f < 6; ++f) {
      qa[f].f4[0] = *(const float4*)(qrow + f * 32 + hi * 8);
      qa[f].f4[1] = *(const float4*)(qrow + f * 32 + 16 + hi * 8);
    }
  }

  v8f acc[8] = {};
  float mstat[8], lstat[8];
#pragma unroll
  for (int r = 0; r < 8; ++r) { mstat[r] = -1e30f; lstat[r] = 0.f; }

  for (int k0 = 0; k0 < t0 + 16; k0 += 32) {
    if (k0 + 32 < t0 + 16) {  // prefetch next chunk's K rows
      __builtin_prefetch(Kf + ((size_t)(k0 + 32 + lane) * NH + h) * QK, 0, 1);
    }
    v8f s[2] = {};
#pragma unroll
    for (int half = 0; half < 2; ++half) {
      const int kt  = k0 + half * 16 + l16;          // this lane's key column
      const int ktc = kt < T_SEQ ? kt : T_SEQ - 1;
      const bf16_t* krow = Kf + ((size_t)ktc * NH + h) * QK;
#pragma unroll
      for (int f = 0; f < 6; ++f) {
        AFragU kbu;
        kbu.f4[0] = *(const float4*)(krow + f * 32 + hi * 8);
        kbu.f4[1] = *(const float4*)(krow + f * 32 + 16 + hi * 8);
        v16bf kb = kbu.v;
        s[half] = __builtin_amdgcn_wmma_f32_16x16x32_bf16(
            false, qa[f].v, false, kb, (short)0, s[half], false, false);
      }
      // causal mask in C layout: VGPR r -> q row t0+r+8*hi, col = kt
#pragma unroll
      for (int r = 0; r < 8; ++r) {
        int trow = t0 + r + hi * 8;
        if (kt > trow) s[half][r] = -1e30f;
      }
    }

    // ---- online softmax (row stats live in registers) ----
    float corr[8];
#pragma unroll
    for (int r = 0; r < 8; ++r) {
      float mx = fmaxf(s[0][r], s[1][r]);
#pragma unroll
      for (int off = 8; off >= 1; off >>= 1)
        mx = fmaxf(mx, __shfl_xor(mx, off, 32));
      float mnew = fmaxf(mstat[r], mx);
      float c = __expf(mstat[r] - mnew);
      mstat[r] = mnew;
      float p0 = __expf(s[0][r] - mnew);
      float p1 = __expf(s[1][r] - mnew);
      s[0][r] = p0; s[1][r] = p1;
      float psum = p0 + p1;
#pragma unroll
      for (int off = 8; off >= 1; off >>= 1)
        psum += __shfl_xor(psum, off, 32);
      lstat[r] = lstat[r] * c + psum;
      corr[r] = c;
    }
#pragma unroll
    for (int j = 0; j < 8; ++j)
#pragma unroll
      for (int r = 0; r < 8; ++r) acc[j][r] *= corr[r];

    // ---- re-stripe P: C layout -> 16x32 row-major in LDS ----
#pragma unroll
    for (int half = 0; half < 2; ++half)
#pragma unroll
      for (int r = 0; r < 8; ++r)
        Ps[(r + hi * 8) * 32 + half * 16 + l16] = (bf16_t)s[half][r];
    __syncthreads();

    AFragU pf;
    pf.f4[0] = *(const float4*)&Ps[l16 * 32 + hi * 8];
    pf.f4[1] = *(const float4*)&Ps[l16 * 32 + 16 + hi * 8];

    // ---- out += P(16x32) * V(32x128): V columns contiguous in Vt ----
#pragma unroll
    for (int j = 0; j < 8; ++j) {
      const bf16_t* vcol =
          Vt + ((size_t)h * VD + j * 16 + l16) * vld + k0;
      AFragU vbu;
      vbu.f4[0] = *(const float4*)(vcol + hi * 8);
      vbu.f4[1] = *(const float4*)(vcol + 16 + hi * 8);
      v16bf vb = vbu.v;
      acc[j] = __builtin_amdgcn_wmma_f32_16x16x32_bf16(
          false, pf.v, false, vb, (short)0, acc[j], false, false);
    }
    __syncthreads();
  }

  // ---- normalize and store O (T, H*VD) bf16 ----
#pragma unroll
  for (int j = 0; j < 8; ++j) {
#pragma unroll
    for (int r = 0; r < 8; ++r) {
      int trow = t0 + r + hi * 8;
      O[(size_t)trow * (NH * VD) + h * VD + j * 16 + l16] =
          (bf16_t)(acc[j][r] / lstat[r]);
    }
  }
}

// ---------------------------------------------------------------------------
// Host orchestration
// ---------------------------------------------------------------------------
extern "C" void kernel_launch(void* const* d_in, const int* in_sizes, int n_in,
                              void* d_out, int out_size, void* d_ws,
                              size_t ws_size, hipStream_t stream) {
  (void)in_sizes; (void)n_in; (void)out_size; (void)ws_size;
  const int*   positions = (const int*)d_in[0];
  const float* hidden    = (const float*)d_in[1];
  const float* w_q_a     = (const float*)d_in[2];
  const float* q_a_norm  = (const float*)d_in[3];
  const float* w_q_b     = (const float*)d_in[4];
  const float* w_kv_a    = (const float*)d_in[5];
  const float* kv_a_norm = (const float*)d_in[6];
  const float* w_kv_b    = (const float*)d_in[7];
  const float* w_o       = (const float*)d_in[8];
  float* out = (float*)d_out;

  char* base = (char*)d_ws;
  size_t off = 0;
  auto carve = [&](size_t bytes) -> char* {
    off = (off + 255) & ~(size_t)255;
    char* p = base + off;
    off += bytes;
    return p;
  };

  const int T = T_SEQ;
  const int VLD = T + 32;  // pad so masked tail reads stay in-bounds
  bf16_t* hs_bf   = (bf16_t*)carve((size_t)T * HID * 2);
  bf16_t* wqa_t   = (bf16_t*)carve((size_t)HID * QL * 2);            // QL x HID
  bf16_t* wqb_t   = (bf16_t*)carve((size_t)QL * NH * QK * 2);        // (H*QK) x QL
  bf16_t* wkva_t  = (bf16_t*)carve((size_t)HID * (KL + ROPE_D) * 2); // 576 x HID
  bf16_t* wkvb_t  = (bf16_t*)carve((size_t)KL * NH * (NOPE + VD) * 2);
  bf16_t* wo_t    = (bf16_t*)carve((size_t)NH * VD * HID * 2);       // HID x 2048
  float*  q_lat   = (float*)carve((size_t)T * QL * 4);
  float*  lat     = (float*)carve((size_t)T * (KL + ROPE_D) * 4);
  bf16_t* q_norm  = (bf16_t*)carve((size_t)T * QL * 2);
  bf16_t* kv_norm = (bf16_t*)carve((size_t)T * KL * 2);
  float*  q_f32   = (float*)carve((size_t)T * NH * QK * 4);
  float*  kv_f32  = (float*)carve((size_t)T * NH * (NOPE + VD) * 4);
  bf16_t* q_full  = (bf16_t*)carve((size_t)T * NH * QK * 2);
  bf16_t* k_full  = (bf16_t*)carve((size_t)T * NH * QK * 2);
  bf16_t* v_t     = (bf16_t*)carve((size_t)NH * VD * VLD * 2);
  bf16_t* attn_o  = (bf16_t*)carve((size_t)T * NH * VD * 2);

  // hidden: plain convert (it is the A operand)
  cvt_bf16_kernel<<<dim3((unsigned)((size_t)T * HID / 1024)), 256, 0, stream>>>(
      hidden, hs_bf, (long)((size_t)T * HID / 4));
  // weights: transpose + convert -> Bt (N x K)
  auto cvt_t = [&](const float* x, bf16_t* y, int Kd, int Nd) {
    cvt_t_bf16_kernel<<<dim3(Nd / 32, Kd / 32), 256, 0, stream>>>(x, y, Kd, Nd);
  };
  cvt_t(w_q_a,  wqa_t,  HID, QL);
  cvt_t(w_q_b,  wqb_t,  QL,  NH * QK);
  cvt_t(w_kv_a, wkva_t, HID, KL + ROPE_D);
  cvt_t(w_kv_b, wkvb_t, KL,  NH * (NOPE + VD));
  cvt_t(w_o,    wo_t,   NH * VD, HID);

  // q_lat = hidden @ w_q_a ; lat = hidden @ w_kv_a
  gemm_bf16_kernel<<<dim3(QL / 64, T / 64), 128, 0, stream>>>(
      hs_bf, wqa_t, q_lat, T, QL, HID);
  gemm_bf16_kernel<<<dim3((KL + ROPE_D) / 64, T / 64), 128, 0, stream>>>(
      hs_bf, wkva_t, lat, T, KL + ROPE_D, HID);

  rmsnorm_bf16_kernel<<<T, 256, 0, stream>>>(q_lat, q_a_norm, q_norm, QL, QL);
  rmsnorm_bf16_kernel<<<T, 256, 0, stream>>>(lat, kv_a_norm, kv_norm, KL,
                                             KL + ROPE_D);

  // q = q_norm @ w_q_b ; kv = kv_norm @ w_kv_b
  gemm_bf16_kernel<<<dim3(NH * QK / 64, T / 64), 128, 0, stream>>>(
      q_norm, wqb_t, q_f32, T, NH * QK, QL);
  gemm_bf16_kernel<<<dim3(NH * (NOPE + VD) / 64, T / 64), 128, 0, stream>>>(
      kv_norm, wkvb_t, kv_f32, T, NH * (NOPE + VD), KL);

  // attn scale = QK^-0.5 * mscale^2 ; cos/sin mscale = 1.0 (mscale==all_dim)
  float mval = 0.1f * logf(40.0f) + 1.0f;
  float attn_scale = powf((float)QK, -0.5f) * mval * mval;
  float csm = 1.0f;

  build_q_kernel<<<T, 256, 0, stream>>>(positions, q_f32, q_full, attn_scale,
                                        csm);
  build_k_kernel<<<T, 256, 0, stream>>>(positions, lat, kv_f32, k_full, v_t,
                                        csm, VLD);

  mla_attn_kernel<<<dim3(T / 16, NH), 32, 0, stream>>>(q_full, k_full, v_t,
                                                       attn_o, VLD);

  // out = attn_o @ w_o  (f32 result straight to d_out)
  gemm_bf16_kernel<<<dim3(HID / 64, T / 64), 128, 0, stream>>>(
      attn_o, wo_t, out, T, HID, NH * VD);
}
